// LegacyGriffinLocalAttention_49890340110994
// MI455X (gfx1250) — matrix operand order repaired and checked by reading, not compiled, not run on hardware
//
#include <hip/hip_runtime.h>
#include <hip/hip_bf16.h>

typedef __attribute__((ext_vector_type(16))) __bf16 v16bf;
typedef __attribute__((ext_vector_type(8)))  __bf16 v8bf;
typedef __attribute__((ext_vector_type(8)))  float  v8f;
typedef unsigned int u32x4 __attribute__((ext_vector_type(4)));
typedef int          i32x8 __attribute__((ext_vector_type(8)));
typedef int          i32x4 __attribute__((ext_vector_type(4)));

#define S_LEN 2048
#define EMB   1024
#define NH    16
#define HD    64
#define WIN   256

static __device__ __forceinline__ v8f wmma_bf16(v16bf a, v16bf b, v8f c) {
  // D = A(16x32 bf16) x B(32x16 bf16) + C(16x16 f32)
  return __builtin_amdgcn_wmma_f32_16x16x32_bf16(false, a, false, b, (short)0, c, false, false);
}

// ---------------- fp32 -> bf16 conversion ----------------
__global__ void cvt_bf16(const float* __restrict__ in, __bf16* __restrict__ out, int n) {
  int i = blockIdx.x * blockDim.x + threadIdx.x;
  if (i < n) out[i] = (__bf16)in[i];
}

// ---------------- GEMM: out = A(bf16 [M,K]) @ W(bf16 [N,K])^T ----------------
// Block: 128 threads = 4 waves. Block tile: 32 rows x 256 cols.
// Wave w handles n0 = blockIdx.y*256 + w*64 (one head's worth of columns).
// A tile (32 x 1024 bf16 = 64KB) staged in LDS via TENSOR_LOAD_TO_LDS (TDM).
// mode 0: Q proj + RoPE -> outb [B,H,S,D]
// mode 1: K proj + RoPE -> outb [B,H,S,D]
// mode 2: V proj        -> outb [B,H,D,S]  (transposed for P*V B-fragments)
// mode 3: O proj        -> outf [B,S,E] fp32
__global__ void proj_kernel(const __bf16* __restrict__ A, const __bf16* __restrict__ W,
                            __bf16* __restrict__ outb, float* __restrict__ outf, int mode)
{
  __shared__ __bf16 Asm[32 * EMB];          // 64 KB A tile
  const int lane = threadIdx.x & 31;
  const int wv   = threadIdx.x >> 5;        // wave in block (0..3)
  const int nl = lane & 15;
  const int hi = lane >> 4;                 // which 16-lane half
  const int m0 = blockIdx.x * 32;           // row tile base (2 x 16)
  const int n0 = blockIdx.y * 256 + wv * 64;// this wave's 64 output cols (one head)

#if __has_builtin(__builtin_amdgcn_tensor_load_to_lds)
  if (wv == 0) {
    // D# group 0: count=1 | lds_addr | global_addr | type=2
    unsigned long long ga = (unsigned long long)(const void*)(A + (size_t)m0 * EMB);
    unsigned lds_base = (unsigned)(unsigned long long)(const void*)(&Asm[0]);
    u32x4 g0 = {};
    g0[0] = 1u;                                   // count = 1 (valid descriptor)
    g0[1] = lds_base;                             // LDS byte address
    g0[2] = (unsigned)(ga & 0xffffffffu);         // global_addr[31:0]
    g0[3] = (unsigned)((ga >> 32) & 0x01ffffffu) | (2u << 30);  // addr[56:32] | type=2
    // D# group 1: data_size=2B, tensor_dim0=1024, tensor_dim1=4096,
    //             tile = 1024 x 32, tensor_dim0_stride = 1024
    i32x8 g1 = {};
    g1[0] = (int)(1u << 16);                      // data_size = 1 (2 bytes)
    g1[1] = (int)(1024u << 16);                   // tensor_dim0[15:0] in bits[63:48]
    g1[2] = (int)(4096u << 16);                   // dim0[31:16]=0, tensor_dim1[15:0]
    g1[3] = (int)(1024u << 16);                   // dim1[31:16]=0, tile_dim0 = 1024
    g1[4] = 32;                                   // tile_dim1 = 32, tile_dim2 = 0
    g1[5] = 1024;                                 // tensor_dim0_stride[31:0]
    g1[6] = 0;                                    // stride hi / dim1_stride lo
    g1[7] = 0;
    i32x4 z4 = {};
    i32x8 z8 = {};
    __builtin_amdgcn_tensor_load_to_lds(g0, g1, z4, z4, z8, 0);
    __builtin_amdgcn_s_wait_tensorcnt(0);         // data resident before barrier
  }
  __syncthreads();
#else
  // cooperative fallback: 128 threads copy 32x1024 bf16
  for (int idx = threadIdx.x; idx < 32 * EMB / 8; idx += 128)
    ((v8bf*)Asm)[idx] = ((const v8bf*)(A + (size_t)m0 * EMB))[idx];
  __syncthreads();
#endif

  const int aoff = hi * 16;                 // K sub-offset for A fragment
  const int boff = hi * 16;                 // K sub-offset for B fragment
  const __bf16* wrow[4];
  #pragma unroll
  for (int j = 0; j < 4; j++) wrow[j] = W + (size_t)(n0 + j * 16 + nl) * EMB;

  union AF { v16bf v; struct { v8bf lo, hii; } p; };
  v8f acc[2][4] = {};
  for (int k0 = 0; k0 < EMB; k0 += 32) {
    __builtin_prefetch(wrow[0] + k0 + 256, 0, 1);
    AF af[2];
    #pragma unroll
    for (int mt = 0; mt < 2; mt++) {
      const __bf16* ar = Asm + (size_t)(mt * 16 + nl) * EMB + k0;   // LDS reads
      af[mt].p.lo  = *(const v8bf*)(ar + aoff);
      af[mt].p.hii = *(const v8bf*)(ar + 8 + aoff);
    }
    #pragma unroll
    for (int j = 0; j < 4; j++) {
      v16bf bf = *(const v16bf*)(wrow[j] + k0 + boff);
      acc[0][j] = wmma_bf16(af[0].v, bf, acc[0][j]);
      acc[1][j] = wmma_bf16(af[1].v, bf, acc[1][j]);
    }
  }

  const int h = n0 >> 6;
  #pragma unroll
  for (int mt = 0; mt < 2; mt++) {
    // RoPE epilogue for Q/K: d and d+32 are column tiles j and j+2 of this wave.
    if (mode <= 1) {
      #pragma unroll
      for (int jp = 0; jp < 2; jp++) {
        int p = jp * 16 + nl;                                   // d % 32
        float invf = __expf(-0.28782313662425572f * (float)p);  // 10000^(-p/32)
        #pragma unroll
        for (int r = 0; r < 8; r++) {
          int gm = m0 + mt * 16 + r + hi * 8;
          int s = gm & (S_LEN - 1);
          float sn, cs;
          __sincosf((float)s * invf, &sn, &cs);
          float lo = acc[mt][jp][r], h2 = acc[mt][jp + 2][r];
          acc[mt][jp][r]     = lo * cs - h2 * sn;               // d < 32
          acc[mt][jp + 2][r] = h2 * cs + lo * sn;               // d >= 32
        }
      }
    }
    #pragma unroll
    for (int r = 0; r < 8; r++) {
      int gm = m0 + mt * 16 + r + hi * 8;          // global row in [0, B*S)
      int b = gm >> 11, s = gm & (S_LEN - 1);
      #pragma unroll
      for (int j = 0; j < 4; j++) {
        int dh = j * 16 + nl;                      // d within head
        float v = acc[mt][j][r];
        if (mode == 3) {
          outf[(size_t)gm * EMB + n0 + dh] = v;
        } else if (mode == 2) {
          outb[(((size_t)(b * NH + h)) * HD + dh) * S_LEN + s] = (__bf16)v;  // V^T
        } else {
          outb[(((size_t)(b * NH + h)) * S_LEN + s) * HD + dh] = (__bf16)v;  // Q/K
        }
      }
    }
  }
}

// ---------------- windowed flash attention: 1 wave = 16 queries of one (b,h) ----------------
__global__ void attn_kernel(const __bf16* __restrict__ Qb, const __bf16* __restrict__ Kb,
                            const __bf16* __restrict__ Vt, __bf16* __restrict__ Yb)
{
  __shared__ __bf16 Plds[16][32];
  const int lane = threadIdx.x & 31;
  const int nl = lane & 15, hi = lane >> 4;
  const int bx = blockIdx.x;
  const int qt = bx & 127;
  const int h  = (bx >> 7) & 15;
  const int b  = bx >> 11;
  const int q0 = qt * 16;
  const size_t bh = (size_t)(b * NH + h);

  const int aoff = hi * 16, boff = hi * 16;
  union AF { v16bf v; struct { v8bf lo, hii; } p; };

  // Q A-fragments for K-dim = 64 (two 16x32 steps)
  const __bf16* qrow = Qb + (bh * S_LEN + q0 + nl) * HD;
  AF aq0, aq1;
  aq0.p.lo  = *(const v8bf*)(qrow + aoff);
  aq0.p.hii = *(const v8bf*)(qrow + 8 + aoff);
  aq1.p.lo  = *(const v8bf*)(qrow + 32 + aoff);
  aq1.p.hii = *(const v8bf*)(qrow + 40 + aoff);

  float mrow[8], lrow[8];
  v8f o[4] = {};
  #pragma unroll
  for (int r = 0; r < 8; r++) { mrow[r] = -1e30f; lrow[r] = 0.f; }

  int t = q0 - (WIN - 1);
  const int kb0 = (t < 0) ? 0 : (t & ~31);
  const int kb1 = (q0 + 15) & ~31;

  for (int kb = kb0; kb <= kb1; kb += 32) {
    // S = Q K^T for 32 keys (two 16x16 tiles)
    v8f s0a = {}, s1a = {};
    const __bf16* k0row = Kb + (bh * S_LEN + kb + nl) * HD;
    const __bf16* k1row = k0row + 16 * HD;
    {
      v16bf b0 = *(const v16bf*)(k0row + boff);
      v16bf b1 = *(const v16bf*)(k0row + 32 + boff);
      s0a = wmma_bf16(aq0.v, b0, s0a);
      s0a = wmma_bf16(aq1.v, b1, s0a);
      v16bf c0 = *(const v16bf*)(k1row + boff);
      v16bf c1 = *(const v16bf*)(k1row + 32 + boff);
      s1a = wmma_bf16(aq0.v, c0, s1a);
      s1a = wmma_bf16(aq1.v, c1, s1a);
    }

    // online softmax per row (row r+hi*8 lives across the 16 lanes of this half)
    #pragma unroll
    for (int r = 0; r < 8; r++) {
      int qpos = q0 + r + hi * 8;
      int rel0 = qpos - (kb + nl);
      int rel1 = rel0 - 16;
      float s0 = (rel0 >= 0 && rel0 < WIN) ? s0a[r] * 0.125f : -1e30f;
      float s1 = (rel1 >= 0 && rel1 < WIN) ? s1a[r] * 0.125f : -1e30f;
      float mx = fmaxf(s0, s1);
      #pragma unroll
      for (int off = 1; off < 16; off <<= 1) mx = fmaxf(mx, __shfl_xor(mx, off, 32));
      float mnew = fmaxf(mrow[r], mx);
      float corr = __expf(mrow[r] - mnew);
      float p0 = __expf(s0 - mnew);
      float p1 = __expf(s1 - mnew);
      float rs = p0 + p1;
      #pragma unroll
      for (int off = 1; off < 16; off <<= 1) rs += __shfl_xor(rs, off, 32);
      lrow[r] = lrow[r] * corr + rs;
      mrow[r] = mnew;
      #pragma unroll
      for (int j = 0; j < 4; j++) o[j][r] *= corr;
      Plds[r + hi * 8][nl]      = (__bf16)p0;
      Plds[r + hi * 8][nl + 16] = (__bf16)p1;
    }
    __syncthreads();

    // O += P(16x32) * V(32x64): transpose P through LDS into A-fragment layout
    AF ap;
    ap.p.lo  = *(const v8bf*)(&Plds[nl][aoff]);
    ap.p.hii = *(const v8bf*)(&Plds[nl][8 + aoff]);
    #pragma unroll
    for (int j = 0; j < 4; j++) {
      const __bf16* vrow = Vt + (bh * HD + j * 16 + nl) * S_LEN + kb + boff;
      v16bf bv = *(const v16bf*)vrow;
      o[j] = wmma_bf16(ap.v, bv, o[j]);
    }
    __syncthreads();
  }

  // normalize and emit bf16 y [B,S,E] with e = h*64 + d
  #pragma unroll
  for (int r = 0; r < 8; r++) {
    int m = r + hi * 8;
    float inv = 1.0f / lrow[r];
    size_t gm = (size_t)b * S_LEN + q0 + m;
    #pragma unroll
    for (int j = 0; j < 4; j++)
      Yb[gm * EMB + h * HD + j * 16 + nl] = (__bf16)(o[j][r] * inv);
  }
}

extern "C" void kernel_launch(void* const* d_in, const int* in_sizes, int n_in,
                              void* d_out, int out_size, void* d_ws, size_t ws_size,
                              hipStream_t stream) {
  (void)in_sizes; (void)n_in; (void)out_size; (void)ws_size;
  const float* x  = (const float*)d_in[0];
  const float* Wq = (const float*)d_in[1];
  const float* Wk = (const float*)d_in[2];
  const float* Wv = (const float*)d_in[3];
  const float* Wo = (const float*)d_in[4];

  const size_t M = 2 * (size_t)S_LEN;          // 4096 rows
  const size_t xN = M * EMB;                   // 4,194,304
  const size_t wN = (size_t)EMB * EMB;         // 1,048,576

  char* ws = (char*)d_ws;
  size_t off = 0;
  __bf16* xb  = (__bf16*)(ws + off); off += xN * 2;
  __bf16* wqb = (__bf16*)(ws + off); off += wN * 2;
  __bf16* wkb = (__bf16*)(ws + off); off += wN * 2;
  __bf16* wvb = (__bf16*)(ws + off); off += wN * 2;
  __bf16* wob = (__bf16*)(ws + off); off += wN * 2;
  __bf16* Qb  = (__bf16*)(ws + off); off += xN * 2;
  __bf16* Kb  = (__bf16*)(ws + off); off += xN * 2;
  __bf16* Vt  = (__bf16*)(ws + off); off += xN * 2;
  __bf16* yb  = (__bf16*)(ws + off); off += xN * 2;

  cvt_bf16<<<(unsigned)((xN + 255) / 256), 256, 0, stream>>>(x,  xb,  (int)xN);
  cvt_bf16<<<(unsigned)((wN + 255) / 256), 256, 0, stream>>>(Wq, wqb, (int)wN);
  cvt_bf16<<<(unsigned)((wN + 255) / 256), 256, 0, stream>>>(Wk, wkb, (int)wN);
  cvt_bf16<<<(unsigned)((wN + 255) / 256), 256, 0, stream>>>(Wv, wvb, (int)wN);
  cvt_bf16<<<(unsigned)((wN + 255) / 256), 256, 0, stream>>>(Wo, wob, (int)wN);

  dim3 gp((unsigned)(M / 32), EMB / 256);      // 128 x 4 blocks, 128 threads (4 waves)
  proj_kernel<<<gp, 128, 0, stream>>>(xb, wqb, Qb, nullptr, 0);
  proj_kernel<<<gp, 128, 0, stream>>>(xb, wkb, Kb, nullptr, 1);
  proj_kernel<<<gp, 128, 0, stream>>>(xb, wvb, Vt, nullptr, 2);

  attn_kernel<<<(unsigned)(2 * NH * (S_LEN / 16)), 32, 0, stream>>>(Qb, Kb, Vt, yb);

  proj_kernel<<<gp, 128, 0, stream>>>(yb, wob, nullptr, (float*)d_out, 3);
}
